// GCNLayer_29987461660874
// MI455X (gfx1250) — matrix operand order repaired and checked by reading, not compiled, not run on hardware
//
#include <hip/hip_runtime.h>
#include <hip/hip_bf16.h>

typedef __attribute__((ext_vector_type(16))) __bf16 v16bf;
typedef __attribute__((ext_vector_type(8)))  __bf16 v8bf;
typedef __attribute__((ext_vector_type(8)))  float  v8f;

union B16x8  { v8bf  v; uint4 u;    };
union B16x16 { v16bf v; uint4 u[2]; };

#define NROW 8192

// -------- pass 1: r[i] = rsqrt(1 + sum_j adj[i,j]) ----------------------
__global__ __launch_bounds__(256) void rowsum_rsqrt_k(const float* __restrict__ adj,
                                                      float* __restrict__ rinv) {
  const int row = blockIdx.x;
  const float* p = adj + (size_t)row * NROW;
  float s = 0.f;
  for (int c = threadIdx.x * 4; c < NROW; c += 256 * 4) {
    float4 v = *(const float4*)(p + c);
    s += v.x + v.y + v.z + v.w;
  }
  __shared__ float red[256];
  red[threadIdx.x] = s;
  __syncthreads();
  for (int off = 128; off > 0; off >>= 1) {
    if (threadIdx.x < off) red[threadIdx.x] += red[threadIdx.x + off];
    __syncthreads();
  }
  if (threadIdx.x == 0) rinv[row] = rsqrtf(red[0] + 1.0f);
}

// Software-pipelined fragment compute: preload 2 B frags, then (wmma j, load j+2).
// sched_group_barrier pins: 6 DS reads; (1 WMMA, 2 DS reads) x (JT-2); 2 WMMA.
template<int JT>
__device__ __forceinline__ void compute_chunk(const unsigned char* ldsA,
                                              const unsigned char* ldsB,
                                              int wr, int wc, int l, v8f (&acc)[JT]) {
  B16x16 afr;
  afr.u[0] = *(const uint4*)(ldsA + wr * 1024 + l * 16);
  afr.u[1] = *(const uint4*)(ldsA + wr * 1024 + 512 + l * 16);
  B16x16 bfr[2];
  bfr[0].u[0] = *(const uint4*)(ldsB + (wc * JT + 0) * 1024 + l * 16);
  bfr[0].u[1] = *(const uint4*)(ldsB + (wc * JT + 0) * 1024 + 512 + l * 16);
  bfr[1].u[0] = *(const uint4*)(ldsB + (wc * JT + 1) * 1024 + l * 16);
  bfr[1].u[1] = *(const uint4*)(ldsB + (wc * JT + 1) * 1024 + 512 + l * 16);
#pragma unroll
  for (int j = 0; j < JT; ++j) {
    acc[j] = __builtin_amdgcn_wmma_f32_16x16x32_bf16(
        false, afr.v, false, bfr[j & 1].v, (short)0, acc[j], false, false);
    if (j + 2 < JT) {
      const int jj = wc * JT + j + 2;
      bfr[j & 1].u[0] = *(const uint4*)(ldsB + jj * 1024 + l * 16);
      bfr[j & 1].u[1] = *(const uint4*)(ldsB + jj * 1024 + 512 + l * 16);
    }
  }
  // Pin the pipeline shape for the scheduler (runs pre-RA, forces 2 live buffers).
  __builtin_amdgcn_sched_group_barrier(0x100, 6, 0);   // A(2) + B0(2) + B1(2)
#pragma unroll
  for (int j = 0; j < JT - 2; ++j) {
    __builtin_amdgcn_sched_group_barrier(0x008, 1, 0); // WMMA j
    __builtin_amdgcn_sched_group_barrier(0x100, 2, 0); // loads for frag j+2
  }
  __builtin_amdgcn_sched_group_barrier(0x008, 2, 0);   // drain
}

// -------- fused WMMA GEMM: C[M x NDIM] = A[M x KDIM] @ B[NDIM x KDIM]^T --
// MODE 0: h1 = x@W1^T + b1                       (B = W1, f32)
// MODE 1: hs = r*(h1@W2^T + b2); store f32 row + bf16 transposed (B = W2, f32)
// MODE 2: out = r*(adj@hs_bf16 + hs_row)         (B = hsT, bf16 N-major; pipelined)
template<int MODE, int JT, int KDIM, int NDIM>
__global__ __launch_bounds__(256)
void gemm_wmma(const float* __restrict__ Af32,
               const void*  __restrict__ Bsrc,
               const float* __restrict__ bias,
               const float* __restrict__ rinv,
               const float* __restrict__ hsrow_in,
               float* __restrict__ Cout,
               __hip_bfloat16* __restrict__ hsT_out) {
  constexpr int TN    = 2 * JT;          // 16x16 tiles along N
  constexpr int NSLOT = (TN * 32) / 256; // B staging slots per thread (2 or 4)
  __shared__ __align__(16) unsigned char lds[4096 + TN * 1024];
  unsigned char* ldsA = lds;             // 4 frags of [2 halves][32 lanes][16B]
  unsigned char* ldsB = lds + 4096;      // TN frags, same shape

  const int t  = threadIdx.x;
  const int l  = t & 31;
  const int w  = t >> 5;
  const int wr = w >> 1;                 // row block 0..3
  const int wc = w & 1;                  // col group 0..1
  const int m0 = blockIdx.x * 64;

  // A staging: thread -> (row ar, k-octet aq); frag addr: lane'=(ar&15)+16*(aq>>1), half=aq&1
  const int ar = t >> 2;
  const int aq = t & 3;
  const int aLdsOff = (ar >> 4) * 1024 + (aq & 1) * 512 + (((ar & 15) + 16 * (aq >> 1)) * 16);
  const float* aG = Af32 + (size_t)(m0 + ar) * KDIM + 8 * aq;

  // B staging slots: lane=(n&15)+16*(k>=16), e=k&15 contiguous 16B halves
  int sn[NSLOT], skoff[NSLOT];
  unsigned char* sdst[NSLOT];
#pragma unroll
  for (int i = 0; i < NSLOT; ++i) {
    const int s  = t + i * 256;
    const int j  = s >> 5;
    const int lb = s & 31;
    sn[i]    = j * 16 + (lb & 15);
    skoff[i] = 16 * (lb >> 4);
    sdst[i]  = ldsB + j * 1024 + lb * 16;
  }

  v8f acc[JT];
#pragma unroll
  for (int j = 0; j < JT; ++j) acc[j] = v8f{};

  if constexpr (MODE == 2) {
    // ------------- register-pipelined main loop (big GEMM) -------------
    const __hip_bfloat16* Bbf = (const __hip_bfloat16*)Bsrc;
    float4 af0, af1;
    uint4  bS[NSLOT][2];
    {
      const float4* p = (const float4*)aG;
      af0 = p[0]; af1 = p[1];
    }
#pragma unroll
    for (int i = 0; i < NSLOT; ++i) {
      const uint4* src = (const uint4*)(Bbf + (size_t)sn[i] * KDIM + skoff[i]);
      bS[i][0] = src[0]; bS[i][1] = src[1];
    }

    for (int kc = 0; kc < KDIM; kc += 32) {
      __syncthreads();
      {
        B16x8 pk;
        pk.v = (v8bf){(__bf16)af0.x, (__bf16)af0.y, (__bf16)af0.z, (__bf16)af0.w,
                      (__bf16)af1.x, (__bf16)af1.y, (__bf16)af1.z, (__bf16)af1.w};
        *(uint4*)(ldsA + aLdsOff) = pk.u;
      }
#pragma unroll
      for (int i = 0; i < NSLOT; ++i) {
        *(uint4*)sdst[i]         = bS[i][0];
        *(uint4*)(sdst[i] + 512) = bS[i][1];
      }
      // issue next chunk's global loads; their wait lands after this chunk's compute
      if (kc + 32 < KDIM) {
        const float4* p = (const float4*)(aG + kc + 32);
        af0 = p[0]; af1 = p[1];
#pragma unroll
        for (int i = 0; i < NSLOT; ++i) {
          const uint4* src = (const uint4*)(Bbf + (size_t)sn[i] * KDIM + (kc + 32) + skoff[i]);
          bS[i][0] = src[0]; bS[i][1] = src[1];
        }
      }
      __syncthreads();
      compute_chunk<JT>(ldsA, ldsB, wr, wc, l, acc);
    }
  } else {
    // ------------- simple loop for the small MLP GEMMs -------------
    for (int kc = 0; kc < KDIM; kc += 32) {
      __syncthreads();
      {
        if (kc + 32 < KDIM) __builtin_prefetch(aG + kc + 32, 0, 3);
        const float4* p = (const float4*)(aG + kc);
        float4 f0 = p[0], f1 = p[1];
        B16x8 pk;
        pk.v = (v8bf){(__bf16)f0.x, (__bf16)f0.y, (__bf16)f0.z, (__bf16)f0.w,
                      (__bf16)f1.x, (__bf16)f1.y, (__bf16)f1.z, (__bf16)f1.w};
        *(uint4*)(ldsA + aLdsOff) = pk.u;
      }
#pragma unroll
      for (int i = 0; i < NSLOT; ++i) {
        const float4* src = (const float4*)((const float*)Bsrc + (size_t)sn[i] * KDIM + kc + skoff[i]);
        float4 g0 = src[0], g1 = src[1], g2 = src[2], g3 = src[3];
        B16x8 p0, p1;
        p0.v = (v8bf){(__bf16)g0.x, (__bf16)g0.y, (__bf16)g0.z, (__bf16)g0.w,
                      (__bf16)g1.x, (__bf16)g1.y, (__bf16)g1.z, (__bf16)g1.w};
        p1.v = (v8bf){(__bf16)g2.x, (__bf16)g2.y, (__bf16)g2.z, (__bf16)g2.w,
                      (__bf16)g3.x, (__bf16)g3.y, (__bf16)g3.z, (__bf16)g3.w};
        *(uint4*)sdst[i]         = p0.u;
        *(uint4*)(sdst[i] + 512) = p1.u;
      }
      __syncthreads();
      compute_chunk<JT>(ldsA, ldsB, wr, wc, l, acc);
    }
  }

  // ---- epilogue (C layout: lane n=l&15, vgpr v -> m = v + 8*(l>=16)) ----
  const int mbase = m0 + wr * 16 + 8 * (l >> 4);
  const int nlane = l & 15;
  float rm[8];
  if constexpr (MODE != 0) {
#pragma unroll
    for (int v = 0; v < 8; ++v) rm[v] = rinv[mbase + v];
  }
#pragma unroll
  for (int j = 0; j < JT; ++j) {
    const int n = (wc * JT + j) * 16 + nlane;
    float bn = 0.f;
    if constexpr (MODE != 2) bn = bias[n];
#pragma unroll
    for (int v = 0; v < 8; ++v) {
      const int m = mbase + v;
      const float val = acc[j][v];
      if constexpr (MODE == 0) {
        Cout[(size_t)m * NDIM + n] = val + bn;
      } else if constexpr (MODE == 1) {
        const float hs = rm[v] * (val + bn);
        Cout[(size_t)m * NDIM + n] = hs;                       // hs row-major f32
        hsT_out[(size_t)n * NROW + m] = __hip_bfloat16(hs);    // hs^T bf16 (GEMM B operand)
      } else {
        Cout[(size_t)m * NDIM + n] = rm[v] * (val + hsrow_in[(size_t)m * NDIM + n]);
      }
    }
  }
}

extern "C" void kernel_launch(void* const* d_in, const int* in_sizes, int n_in,
                              void* d_out, int out_size, void* d_ws, size_t ws_size,
                              hipStream_t stream) {
  const float* x   = (const float*)d_in[0];   // [8192,512]
  const float* adj = (const float*)d_in[1];   // [8192,8192]
  const float* W1  = (const float*)d_in[2];   // [256,512]
  const float* b1  = (const float*)d_in[3];   // [256]
  const float* W2  = (const float*)d_in[4];   // [512,256]
  const float* b2  = (const float*)d_in[5];   // [512]
  float* out = (float*)d_out;                 // [8192,512]

  char* ws = (char*)d_ws;
  float* rinv         = (float*)ws;                                        // 32 KB
  float* h1           = (float*)(ws + (1u << 15));                         // 8 MB
  float* hsrow        = (float*)(ws + (1u << 15) + (8u << 20));            // 16 MB
  __hip_bfloat16* hsT = (__hip_bfloat16*)(ws + (1u << 15) + (24u << 20));  // 8 MB

  rowsum_rsqrt_k<<<NROW, 256, 0, stream>>>(adj, rinv);
  // h1 = x @ W1^T + b1        : M=8192 N=256 K=512
  gemm_wmma<0, 8, 512, 256><<<NROW / 64, 256, 0, stream>>>(
      x, (const void*)W1, b1, nullptr, nullptr, h1, nullptr);
  // hs = r * (h1 @ W2^T + b2) : M=8192 N=512 K=256 ; writes f32 rows + bf16 transpose
  gemm_wmma<1, 16, 256, 512><<<NROW / 64, 256, 0, stream>>>(
      h1, (const void*)W2, b2, rinv, nullptr, hsrow, hsT);
  // out = r * (adj @ hs + hs) : M=8192 N=512 K=8192, bf16 B operand
  gemm_wmma<2, 16, 8192, 512><<<NROW / 64, 256, 0, stream>>>(
      adj, (const void*)hsT, nullptr, rinv, hsrow, out, nullptr);
}
